// HybridGemmaDiT_33165737459847
// MI455X (gfx1250) — matrix operand order repaired and checked by reading, not compiled, not run on hardware
//
#include <hip/hip_runtime.h>
#include <math.h>

// ---------------------------------------------------------------------------
// Types for CDNA5 WMMA
// ---------------------------------------------------------------------------
typedef __attribute__((ext_vector_type(16))) _Float16 v16h;
typedef __attribute__((ext_vector_type(8)))  _Float16 v8h;
typedef __attribute__((ext_vector_type(8)))  float    v8f;

#define WMMA_F16(a, b, c) \
    __builtin_amdgcn_wmma_f32_16x16x32_f16(false, (a), false, (b), (short)0, (c), false, false)

#define CAT16(lo, hi) __builtin_shufflevector((lo), (hi), 0,1,2,3,4,5,6,7,8,9,10,11,12,13,14,15)

// Model constants
#define BD   64           // batch
#define SD   256          // tokens
#define DD   768          // model dim
#define NHD  4            // heads
#define HDD  192          // head dim
#define LD   8            // layers
#define NTOK 16384        // B*S
#define EPSF 1.1920929e-07f

__device__ __forceinline__ unsigned lds_off(const void* p) {
    return (unsigned)(unsigned long long)p;   // flat addr[31:0] == LDS offset
}
__device__ __forceinline__ void async_b128(unsigned ldsoff, const _Float16* gp) {
    asm volatile("global_load_async_to_lds_b128 %0, %1, off"
                 :: "v"(ldsoff), "v"((unsigned long long)gp) : "memory");
}
__device__ __forceinline__ void wait_async0() {
    asm volatile("s_wait_asynccnt 0" ::: "memory");
}

// ---------------------------------------------------------------------------
// GEMM: C[M,N] = A[M,K] @ B[K,N] (+ bias[n]).
// A: f32 row-major (lda). B given PRE-TRANSPOSED f16: BT[n][k], row stride K.
// Block = 128 threads (4 waves) -> 128x64 C tile; wave w owns rows
// [w*32, w*32+32) as two 16-row WMMA fragments (8 accumulators).
// Double-buffered LDS pipeline: next A tile prefetched to registers + next B
// tile DMA'd with global_load_async_to_lds_b128 (ASYNCcnt) while current tile
// computes; one barrier per K-step. All fragments loaded before the WMMA
// burst so waits coalesce. Requires M%128==0, N%64==0, K%32==0.
// ---------------------------------------------------------------------------
__global__ void __launch_bounds__(128)
gemm128x64(const float* __restrict__ A, const _Float16* __restrict__ BT,
           const float* __restrict__ bias, float* __restrict__ C,
           int M, int N, int K, int lda, int ldc)
{
    __shared__ _Float16 As[2][128 * 32];   // [m][k]
    __shared__ _Float16 Bs[2][64 * 32];    // [n][k]
    const int t    = threadIdx.x;
    const int lane = t & 31;
    const int w    = t >> 5;
    const int m0 = blockIdx.y << 7;
    const int n0 = blockIdx.x << 6;

    const int mA  = lane & 15;
    const int hiA = (lane >> 4) << 4;    // A K-offset: +16 for upper half-wave
    const int nB  = lane & 15;
    const int hiB = (lane >> 4) << 4;    // B K-offset: +16 for upper half-wave

    // staging coordinates (uniform trip counts, no divergence)
    const int ar0  = t >> 2;             // A rows: ar0 + 32*i, i=0..3
    const int aseg = (t & 3) << 3;       // 8-float segment within 32-wide row
    const int br0  = t >> 2;             // B rows: br0 + 32*i, i=0..1
    const int bseg = (t & 3) << 3;

    v8f acc[8] = {{}, {}, {}, {}, {}, {}, {}, {}};

    // ---- prologue: stage k0 = 0 into buffer 0 ----
#pragma unroll
    for (int i = 0; i < 4; ++i) {
        const int ar = ar0 + 32 * i;
        const float* ap = A + (long)(m0 + ar) * lda + aseg;
        v8h h;
#pragma unroll
        for (int j = 0; j < 8; ++j) h[j] = (_Float16)ap[j];
        *(v8h*)(&As[0][ar * 32 + aseg]) = h;
    }
#pragma unroll
    for (int i = 0; i < 2; ++i) {
        const int br = br0 + 32 * i;
        async_b128(lds_off(&Bs[0][br * 32 + bseg]),
                   BT + (long)(n0 + br) * K + bseg);
    }
    wait_async0();
    __syncthreads();

    int buf = 0;
    for (int k0 = 0; k0 < K; k0 += 32) {
        const int kn = k0 + 32;
        const bool has_next = kn < K;
        float af[4][8];
        if (has_next) {
            // prefetch next A tile into registers (global_load_b128 pairs)
#pragma unroll
            for (int i = 0; i < 4; ++i) {
                const float* ap = A + (long)(m0 + ar0 + 32 * i) * lda + kn + aseg;
#pragma unroll
                for (int j = 0; j < 8; ++j) af[i][j] = ap[j];
            }
            // next B tile: async DMA straight into the other LDS buffer
#pragma unroll
            for (int i = 0; i < 2; ++i) {
                const int br = br0 + 32 * i;
                async_b128(lds_off(&Bs[buf ^ 1][br * 32 + bseg]),
                           BT + (long)(n0 + br) * K + kn + bseg);
            }
        }
        // ---- load all fragments, then issue the WMMA burst ----
        const int arow = (w << 5) + mA;
        v8h l0 = *(const v8h*)(&As[buf][arow * 32 + hiA]);
        v8h h0 = *(const v8h*)(&As[buf][arow * 32 + hiA + 8]);
        v8h l1 = *(const v8h*)(&As[buf][(arow + 16) * 32 + hiA]);
        v8h h1 = *(const v8h*)(&As[buf][(arow + 16) * 32 + hiA + 8]);
        v16h a0 = CAT16(l0, h0);
        v16h a1 = CAT16(l1, h1);
        v16h bb[4];
#pragma unroll
        for (int tn = 0; tn < 4; ++tn) {
            const int brow = (tn << 4) + nB;
            v8h bl = *(const v8h*)(&Bs[buf][brow * 32 + hiB]);
            v8h bh = *(const v8h*)(&Bs[buf][brow * 32 + hiB + 8]);
            bb[tn] = CAT16(bl, bh);
        }
#pragma unroll
        for (int tn = 0; tn < 4; ++tn) {
            acc[tn]     = WMMA_F16(a0, bb[tn], acc[tn]);
            acc[4 + tn] = WMMA_F16(a1, bb[tn], acc[4 + tn]);
        }
        // ---- convert + store prefetched A into the other buffer ----
        if (has_next) {
#pragma unroll
            for (int i = 0; i < 4; ++i) {
                v8h h;
#pragma unroll
                for (int j = 0; j < 8; ++j) h[j] = (_Float16)af[i][j];
                *(v8h*)(&As[buf ^ 1][(ar0 + 32 * i) * 32 + aseg]) = h;
            }
        }
        wait_async0();
        __syncthreads();
        buf ^= 1;
    }

    const int mbase = m0 + (w << 5) + ((lane >> 4) << 3);
#pragma unroll
    for (int sub = 0; sub < 2; ++sub) {
#pragma unroll
        for (int tn = 0; tn < 4; ++tn) {
            const int n = n0 + (tn << 4) + nB;
            const float e = bias ? bias[n] : 0.f;
#pragma unroll
            for (int r = 0; r < 8; ++r)
                C[(long)(mbase + sub * 16 + r) * ldc + n] = acc[sub * 4 + tn][r] + e;
        }
    }
}

// ---------------------------------------------------------------------------
// Weight prep: out[n*K + k] = (f16) in[k*N + n]   (transpose + f32->f16)
// ---------------------------------------------------------------------------
__global__ void __launch_bounds__(256)
w_to_f16t_kernel(const float* __restrict__ in, _Float16* __restrict__ out,
                 int K, int N, long total)
{
    const long idx = (long)blockIdx.x * 256 + threadIdx.x;
    if (idx >= total) return;
    const int  k = (int)(idx % K);
    const long n = idx / K;
    out[idx] = (_Float16)in[(long)k * N + n];
}

// ---------------------------------------------------------------------------
// Attention: one wave per (query-tile-of-16, head, batch).
// scores = (q @ k^T) * scale, masked (local 2.5-radius or global layer),
// softmax over 256 keys in LDS (f32), then P @ V. All matmuls via WMMA.
// q/k/v/ctx layout: [B*S, 768] row-major, head h at columns h*192..h*192+191.
// ---------------------------------------------------------------------------
__global__ void __launch_bounds__(32)
attn_wmma(const float* __restrict__ q, const float* __restrict__ k,
          const float* __restrict__ v, float* __restrict__ ctx, int glob)
{
    __shared__ float sc[16][SD];   // 16 KB
    const int lane = threadIdx.x;
    const int qt = blockIdx.x, h = blockIdx.y, b = blockIdx.z;
    const long base = (long)b * SD * DD + h * HDD;
    const int mA  = lane & 15;
    const int hiA = (lane >> 4) << 4;
    const int nB  = lane & 15;
    const int hiB = (lane >> 4) << 4;

    v16h qa[6];
#pragma unroll
    for (int c = 0; c < 6; ++c) {
        const float* ap = q + base + (long)(qt * 16 + mA) * DD + c * 32 + hiA;
#pragma unroll
        for (int j = 0; j < 16; ++j)
            qa[c][j] = (_Float16)ap[j];
    }

    for (int kt = 0; kt < 16; ++kt) {
        v8f acc = {};
#pragma unroll
        for (int c = 0; c < 6; ++c) {
            const float* bp = k + base + (long)(kt * 16 + nB) * DD + c * 32 + hiB;
            v16h kb;
#pragma unroll
            for (int j = 0; j < 16; ++j) kb[j] = (_Float16)bp[j];
            acc = WMMA_F16(qa[c], kb, acc);
        }
        const int key = kt * 16 + nB;
        const int ky = key >> 4, kx = key & 15;
#pragma unroll
        for (int r = 0; r < 8; ++r) {
            const int m  = ((lane >> 4) << 3) + r;
            const int qi = qt * 16 + m;
            const int dy = (qi >> 4) - ky, dx = (qi & 15) - kx;
            const bool keep = (glob != 0) || ((float)(dy * dy + dx * dx) < 6.25f);
            sc[m][key] = keep ? acc[r] * 0.07216878364870323f : -1e30f;  // 1/sqrt(192)
        }
    }
    __syncthreads();

    if (lane < 16) {
        float mx = -3.0e38f;
        for (int j = 0; j < SD; ++j) mx = fmaxf(mx, sc[lane][j]);
        float sum = 0.f;
        for (int j = 0; j < SD; ++j) { float e = expf(sc[lane][j] - mx); sc[lane][j] = e; sum += e; }
        float inv = 1.f / sum;
        for (int j = 0; j < SD; ++j) sc[lane][j] *= inv;
    }
    __syncthreads();

    for (int ct = 0; ct < 12; ++ct) {
        v8f acc = {};
        for (int kc = 0; kc < 8; ++kc) {
            v16h pa;
            const float* ap = &sc[mA][kc * 32 + hiA];
#pragma unroll
            for (int j = 0; j < 16; ++j)
                pa[j] = (_Float16)ap[j];
            v16h vb;
            const float* bp = v + base + (long)(kc * 32 + hiB) * DD + ct * 16 + nB;
#pragma unroll
            for (int j = 0; j < 16; ++j) vb[j] = (_Float16)bp[(long)j * DD];
            acc = WMMA_F16(pa, vb, acc);
        }
#pragma unroll
        for (int r = 0; r < 8; ++r) {
            const int m = ((lane >> 4) << 3) + r;
            ctx[base + (long)(qt * 16 + m) * DD + ct * 16 + nB] = acc[r];
        }
    }
}

// ---------------------------------------------------------------------------
// Householder scan (exactly mirrors reference lax.scan). One 256-thread block
// per layer; f32 Q kept in global scratch, fenced between phases.
// Emits f16 Q (row-major) and f16 Q^T (row-major) for the GEMM B^T inputs.
// ---------------------------------------------------------------------------
__global__ void __launch_bounds__(256)
householder_kernel(const float* __restrict__ hh_vs, float* __restrict__ Qsc,
                   _Float16* __restrict__ Qh, _Float16* __restrict__ QTh)
{
    const int l = blockIdx.x;
    float* Q = Qsc + (long)l * HDD * HDD;
    __shared__ float v[HDD], w[HDD], red[256];
    const int t = threadIdx.x;

    for (int i = t; i < HDD * HDD; i += 256)
        Q[i] = ((i / HDD) == (i % HDD)) ? 1.f : 0.f;
    __threadfence(); __syncthreads();

    for (int s = 0; s < HDD / 2; ++s) {
        const float* vs = hh_vs + ((long)l * (HDD / 2) + s) * HDD;
        if (t < HDD) v[t] = vs[t];
        __syncthreads();
        float p = (t < HDD) ? v[t] * v[t] : 0.f;
        red[t] = p; __syncthreads();
        for (int o = 128; o > 0; o >>= 1) { if (t < o) red[t] += red[t + o]; __syncthreads(); }
        const float c = 2.f / (red[0] + 1e-8f);
        __syncthreads();
        if (t < HDD) {
            float acc = 0.f;
            for (int i = 0; i < HDD; ++i) acc += v[i] * Q[i * HDD + t];
            w[t] = acc;
        }
        __syncthreads();
        for (int i = t; i < HDD * HDD; i += 256) {
            const int r = i / HDD, cc = i % HDD;
            Q[i] -= c * v[r] * w[cc];
        }
        __threadfence(); __syncthreads();
    }
    for (int i = t; i < HDD * HDD; i += 256) {
        const float qv = Q[i];
        Qh [(long)l * HDD * HDD + i] = (_Float16)qv;
        QTh[(long)l * HDD * HDD + (i % HDD) * HDD + (i / HDD)] = (_Float16)qv;
    }
}

// ---------------------------------------------------------------------------
// RMSNorm: one block per row of [NTOK, 768]
// ---------------------------------------------------------------------------
__global__ void __launch_bounds__(256)
rmsnorm_kernel(const float* __restrict__ x, float* __restrict__ y)
{
    __shared__ float red[256];
    const long row = blockIdx.x;
    const float* xr = x + row * DD;
    float* yr = y + row * DD;
    const int t = threadIdx.x;
    float s = 0.f;
    for (int i = t; i < DD; i += 256) { float v = xr[i]; s += v * v; }
    red[t] = s; __syncthreads();
    for (int o = 128; o > 0; o >>= 1) { if (t < o) red[t] += red[t + o]; __syncthreads(); }
    const float r = rsqrtf(red[0] * (1.f / DD) + EPSF);
    for (int i = t; i < DD; i += 256) yr[i] = xr[i] * r;
}

// ---------------------------------------------------------------------------
// Patchify + timestep embedding
// ---------------------------------------------------------------------------
__global__ void __launch_bounds__(256)
embed_kernel(const float* __restrict__ z, const float* __restrict__ logsnr,
             const float* __restrict__ pw, const float* __restrict__ pb,
             float* __restrict__ x)
{
    const long idx = (long)blockIdx.x * 256 + threadIdx.x;
    if (idx >= (long)NTOK * DD) return;
    const int d = (int)(idx % DD);
    const long bs = idx / DD;
    const int s = (int)(bs % SD), b = (int)(bs / SD);
    const int gy = s >> 4, gx = s & 15;
    float acc = pb[d];
#pragma unroll
    for (int c = 0; c < 3; ++c)
#pragma unroll
        for (int dy = 0; dy < 2; ++dy)
#pragma unroll
            for (int dx = 0; dx < 2; ++dx) {
                const float pv = z[(((long)b * 3 + c) * 32 + (gy * 2 + dy)) * 32 + (gx * 2 + dx)];
                acc += pv * pw[(c * 4 + dy * 2 + dx) * DD + d];
            }
    const float ls = logsnr[b];
#pragma unroll
    for (int i = 0; i < 4; ++i) {
        const float ang = ls * (float)(1 << i);
        acc += sinf(ang) * pw[(12 + i) * DD + d] + cosf(ang) * pw[(16 + i) * DD + d];
    }
    x[idx] = acc;
}

// ---------------------------------------------------------------------------
// RoPE (2D axial): in-place on t[65536, 192]; row = (b*256+s)*4 + h.
// ---------------------------------------------------------------------------
__global__ void __launch_bounds__(256)
rope_kernel(float* __restrict__ tt)
{
    const long idx = (long)blockIdx.x * 256 + threadIdx.x;
    if (idx >= (long)NTOK * NHD * 96) return;
    const int d = (int)(idx % 96);
    const long row = idx / 96;
    const int s = (int)((row >> 2) & 255);
    const float coord = (d < 48) ? (float)(s >> 4) : (float)(s & 15);
    const int i = d % 48;
    const float angle = coord * powf(10000.f, -(float)i / 48.f);
    const float cs = cosf(angle), sn = sinf(angle);
    float* p = tt + row * HDD;
    const float a = p[d], b = p[d + 96];
    p[d]      = a * cs - b * sn;
    p[d + 96] = b * cs + a * sn;
}

// ---------------------------------------------------------------------------
// Elementwise epilogues
// ---------------------------------------------------------------------------
__global__ void __launch_bounds__(256)
gate_res_kernel(float* __restrict__ x, const float* __restrict__ a,
                const float* __restrict__ g, long n)
{
    const long i = (long)blockIdx.x * 256 + threadIdx.x;
    if (i < n) x[i] += a[i] * (1.f / (1.f + expf(-g[i])));
}

__global__ void __launch_bounds__(256)
mlp_act_kernel(float* __restrict__ g, const float* __restrict__ val, long n)
{
    const long i = (long)blockIdx.x * 256 + threadIdx.x;
    if (i < n) {
        const float gv = g[i];
        g[i] = val[i] * (0.5f * gv * (1.f + erff(gv * 0.7071067811865475f)));
    }
}

__global__ void __launch_bounds__(256)
add_kernel(float* __restrict__ x, const float* __restrict__ y, long n)
{
    const long i = (long)blockIdx.x * 256 + threadIdx.x;
    if (i < n) x[i] += y[i];
}

// ---------------------------------------------------------------------------
// scale_f[b, 0..11] = exp(silu(spins @ sd1_w + sd1_b) @ sd2_w + sd2_b)
// ---------------------------------------------------------------------------
__global__ void __launch_bounds__(256)
scalef_kernel(const float* __restrict__ logsnr, const float* __restrict__ sd1w,
              const float* __restrict__ sd1b, const float* __restrict__ sd2w,
              const float* __restrict__ sd2b, float* __restrict__ scalef)
{
    __shared__ float sv[DD];
    const int b = blockIdx.x, t = threadIdx.x;
    const float ls = logsnr[b];
    float sp[8];
#pragma unroll
    for (int i = 0; i < 4; ++i) { const float a = ls * (float)(1 << i); sp[i] = sinf(a); sp[4 + i] = cosf(a); }
    for (int d = t; d < DD; d += 256) {
        float acc = sd1b[d];
#pragma unroll
        for (int j = 0; j < 8; ++j) acc += sp[j] * sd1w[j * DD + d];
        sv[d] = acc / (1.f + expf(-acc));   // silu
    }
    __syncthreads();
    if (t < 12) {
        float acc = sd2b[t];
        for (int d = 0; d < DD; ++d) acc += sv[d] * sd2w[d * 12 + t];
        scalef[b * 12 + t] = expf(acc);
    }
}

// ---------------------------------------------------------------------------
// Head + per-channel scale + unpatchify into d_out (z_img portion)
// ---------------------------------------------------------------------------
__global__ void __launch_bounds__(256)
head_kernel(const float* __restrict__ h, const float* __restrict__ hw,
            const float* __restrict__ hb, const float* __restrict__ scalef,
            float* __restrict__ out)
{
    const long idx = (long)blockIdx.x * 256 + threadIdx.x;
    if (idx >= (long)NTOK * 12) return;
    const int j = (int)(idx % 12);
    const long bs = idx / 12;
    const int s = (int)(bs % SD), b = (int)(bs / SD);
    const float* hr = h + bs * DD;
    float acc = hb[j];
    for (int d = 0; d < DD; ++d) acc += hr[d] * hw[d * 12 + j];
    acc *= scalef[b * 12 + j];
    const int c = j >> 2, dy = (j >> 1) & 1, dx = j & 1;
    const int gy = s >> 4, gx = s & 15;
    out[(((long)b * 3 + c) * 32 + (gy * 2 + dy)) * 32 + (gx * 2 + dx)] = acc;
}

// ---------------------------------------------------------------------------
// l_pred[b] = mean_s(x_norm[b,s,:]) @ lam_w + lam_b
// ---------------------------------------------------------------------------
__global__ void __launch_bounds__(256)
lam_kernel(const float* __restrict__ x, const float* __restrict__ lamw,
           const float* __restrict__ lamb, float* __restrict__ out)
{
    __shared__ float red[256];
    const int b = blockIdx.x, t = threadIdx.x;
    float acc = 0.f;
    for (int d = t; d < DD; d += 256) {
        float sum = 0.f;
        for (int s = 0; s < SD; ++s) sum += x[((long)b * SD + s) * DD + d];
        acc += (sum * (1.f / SD)) * lamw[d];
    }
    red[t] = acc; __syncthreads();
    for (int o = 128; o > 0; o >>= 1) { if (t < o) red[t] += red[t + o]; __syncthreads(); }
    if (t == 0) out[(long)BD * 3 * 32 * 32 + b] = red[0] + lamb[0];
}

// ---------------------------------------------------------------------------
// Host-side launch
// ---------------------------------------------------------------------------
static inline void gemm(hipStream_t st, const float* A, const _Float16* BT, const float* bias,
                        float* C, int M, int N, int K, int lda, int ldc)
{
    dim3 g((unsigned)(N / 64), (unsigned)(M / 128));
    gemm128x64<<<g, 128, 0, st>>>(A, BT, bias, C, M, N, K, lda, ldc);
}

static inline void wconv(hipStream_t st, const float* in, _Float16* out, int K, int N)
{
    const long total = (long)K * N;
    w_to_f16t_kernel<<<(int)((total + 255) / 256), 256, 0, st>>>(in, out, K, N, total);
}

extern "C" void kernel_launch(void* const* d_in, const int* in_sizes, int n_in,
                              void* d_out, int out_size, void* d_ws, size_t ws_size,
                              hipStream_t stream)
{
    (void)in_sizes; (void)n_in; (void)out_size; (void)ws_size;
    const float* z_t    = (const float*)d_in[0];
    const float* logsnr = (const float*)d_in[1];
    const float* qkvw   = (const float*)d_in[2];
    const float* outw   = (const float*)d_in[3];
    const float* gatew  = (const float*)d_in[4];
    const float* gateb  = (const float*)d_in[5];
    const float* mlpgw  = (const float*)d_in[6];
    const float* mlpgb  = (const float*)d_in[7];
    const float* mlpow  = (const float*)d_in[8];
    const float* mlpob  = (const float*)d_in[9];
    const float* hhvs   = (const float*)d_in[10];
    const float* pw     = (const float*)d_in[11];
    const float* pb     = (const float*)d_in[12];
    const float* headw  = (const float*)d_in[13];
    const float* headb  = (const float*)d_in[14];
    const float* sd1w   = (const float*)d_in[15];
    const float* sd1b   = (const float*)d_in[16];
    const float* sd2w   = (const float*)d_in[17];
    const float* sd2b   = (const float*)d_in[18];
    const float* lamw   = (const float*)d_in[19];
    const float* lamb   = (const float*)d_in[20];
    float* out = (float*)d_out;

    // ---- workspace layout ----
    const long ND = (long)NTOK * DD;          // 12,582,912
    const long NM = (long)NTOK * 4 * DD;      // 50,331,648
    const long QQ = (long)LD * HDD * HDD;     // 294,912
    float* ws   = (float*)d_ws;
    float* xbuf = ws;
    float* hbuf = xbuf + ND;
    float* qbuf = hbuf + ND;
    float* kbuf = qbuf + ND;
    float* vbuf = kbuf + ND;
    float* cbuf = vbuf + ND;
    float* dbuf = cbuf + ND;
    float* gbuf = dbuf + ND;                  // [16384, 3072]
    float* vlbf = gbuf + NM;                  // [16384, 3072]
    float* qsc  = vlbf + NM;                  // f32 Householder scratch
    float* scf  = qsc + QQ;                   // 64 x 12
    // f16 region (pre-transposed weights: W^T[n][k])
    _Float16* f16 = (_Float16*)(scf + 1024);
    _Float16* qkvw16 = f16;                                     // 8 x 2304 x 768
    _Float16* outw16 = qkvw16 + (long)LD * 3 * DD * DD;         // 8 x 768 x 768
    _Float16* gatw16 = outw16 + (long)LD * DD * DD;
    _Float16* mgw16  = gatw16 + (long)LD * DD * DD;             // 8 x 6144 x 768
    _Float16* mow16  = mgw16 + (long)LD * 8 * DD * DD;          // 8 x 768 x 3072
    _Float16* Qh     = mow16 + (long)LD * 4 * DD * DD;
    _Float16* QTh    = Qh + QQ;

    const long EW = ND;
    const int  EB = (int)((EW + 255) / 256);
    const int  MB = (int)((NM + 255) / 256);

    const long QKVW_L = (long)DD * 3 * DD;
    const long OW_L   = (long)DD * DD;
    const long MGW_L  = (long)DD * 8 * DD;
    const long MOW_L  = (long)4 * DD * DD;

    // ---- weight prep: transpose + f32->f16 ----
    for (int l = 0; l < LD; ++l) {
        wconv(stream, qkvw + l * QKVW_L, qkvw16 + l * QKVW_L, DD, 3 * DD);
        wconv(stream, outw + l * OW_L,   outw16 + l * OW_L,   DD, DD);
        wconv(stream, gatew + l * OW_L,  gatw16 + l * OW_L,   DD, DD);
        wconv(stream, mlpgw + l * MGW_L, mgw16 + l * MGW_L,   DD, 8 * DD);
        wconv(stream, mlpow + l * MOW_L, mow16 + l * MOW_L,   4 * DD, DD);
    }
    householder_kernel<<<LD, 256, 0, stream>>>(hhvs, qsc, Qh, QTh);
    embed_kernel<<<EB, 256, 0, stream>>>(z_t, logsnr, pw, pb, xbuf);
    scalef_kernel<<<BD, 256, 0, stream>>>(logsnr, sd1w, sd1b, sd2w, sd2b, scf);

    for (int l = 0; l < LD; ++l) {
        const int glob = ((l + 1) % 4 == 0) ? 1 : 0;
        // h = rmsnorm(x); q,k,v = h @ qkv_w  (B^T row slices: q rows 0..767, ...)
        rmsnorm_kernel<<<NTOK, 256, 0, stream>>>(xbuf, hbuf);
        gemm(stream, hbuf, qkvw16 + l * QKVW_L,              nullptr, qbuf, NTOK, DD, DD, DD, DD);
        gemm(stream, hbuf, qkvw16 + l * QKVW_L + OW_L,       nullptr, kbuf, NTOK, DD, DD, DD, DD);
        gemm(stream, hbuf, qkvw16 + l * QKVW_L + 2 * OW_L,   nullptr, vbuf, NTOK, DD, DD, DD, DD);
        // q = rot(q @ Q^T) @ Q   (B = Q^T -> B^T = Q; B = Q -> B^T = Q^T)
        gemm(stream, qbuf, Qh + (long)l * HDD * HDD, nullptr, hbuf, NTOK * NHD, HDD, HDD, HDD, HDD);
        rope_kernel<<<(int)(((long)NTOK * NHD * 96 + 255) / 256), 256, 0, stream>>>(hbuf);
        gemm(stream, hbuf, QTh + (long)l * HDD * HDD, nullptr, qbuf, NTOK * NHD, HDD, HDD, HDD, HDD);
        // k = rot(k @ Q^T) @ Q
        gemm(stream, kbuf, Qh + (long)l * HDD * HDD, nullptr, hbuf, NTOK * NHD, HDD, HDD, HDD, HDD);
        rope_kernel<<<(int)(((long)NTOK * NHD * 96 + 255) / 256), 256, 0, stream>>>(hbuf);
        gemm(stream, hbuf, QTh + (long)l * HDD * HDD, nullptr, kbuf, NTOK * NHD, HDD, HDD, HDD, HDD);
        // attention
        attn_wmma<<<dim3(SD / 16, NHD, BD), 32, 0, stream>>>(qbuf, kbuf, vbuf, cbuf, glob);
        // out projection, gate, residual
        gemm(stream, cbuf, outw16 + l * OW_L, nullptr, dbuf, NTOK, DD, DD, DD, DD);
        gemm(stream, dbuf, gatw16 + l * OW_L, gateb + (long)l * DD, hbuf, NTOK, DD, DD, DD, DD);
        gate_res_kernel<<<EB, 256, 0, stream>>>(xbuf, dbuf, hbuf, EW);
        // MLP
        rmsnorm_kernel<<<NTOK, 256, 0, stream>>>(xbuf, hbuf);
        gemm(stream, hbuf, mgw16 + l * MGW_L,                 mlpgb + (long)l * 8 * DD,          gbuf, NTOK, 4 * DD, DD, DD, 4 * DD);
        gemm(stream, hbuf, mgw16 + l * MGW_L + (long)4*DD*DD, mlpgb + (long)l * 8 * DD + 4 * DD, vlbf, NTOK, 4 * DD, DD, DD, 4 * DD);
        mlp_act_kernel<<<MB, 256, 0, stream>>>(gbuf, vlbf, NM);
        gemm(stream, gbuf, mow16 + l * MOW_L, mlpob + (long)l * DD, dbuf, NTOK, DD, 4 * DD, 4 * DD, DD);
        add_kernel<<<EB, 256, 0, stream>>>(xbuf, dbuf, EW);
    }

    // final norm, head (+scale, unpatchify), lambda head
    rmsnorm_kernel<<<NTOK, 256, 0, stream>>>(xbuf, hbuf);
    head_kernel<<<(int)(((long)NTOK * 12 + 255) / 256), 256, 0, stream>>>(hbuf, headw, headb, scf, out);
    lam_kernel<<<BD, 256, 0, stream>>>(hbuf, lamw, lamb, out);
}
